// PointerNetwork_73048803770834
// MI455X (gfx1250) — matrix operand order repaired and checked by reading, not compiled, not run on hardware
//
#include <hip/hip_runtime.h>
#include <hip/hip_bf16.h>

// ---------------------------------------------------------------------------
// CDNA5 (gfx1250) PointerNetwork: encoder LSTM + enc_proj + decoder attention.
// All GEMMs via v_wmma_f32_16x16x32_bf16 (wave32, fp32 accumulate).
// ---------------------------------------------------------------------------

typedef __attribute__((ext_vector_type(16))) __bf16 v16bf;
typedef __attribute__((ext_vector_type(8)))  __bf16 v8bf;
typedef __attribute__((ext_vector_type(8)))  float  v8f;

// ---- WMMA fragment loads (layouts per CDNA5 ISA §7.12.2) --------

// A matrix 16x32 bf16, row m striped across a lane's 16 elements:
// lane in [0,15]  : M=lane,   K = {k0+0..7,  k0+16..23}
// lane in [16,31] : M=lane-16,K = {k0+8..15, k0+24..31}
__device__ __forceinline__ v16bf load_frag_a(const __bf16* base, int lda,
                                             int m0, int k0) {
  const int lane = threadIdx.x & 31;
  const int hf   = lane >> 4;
  const __bf16* row = base + (long)(m0 + (lane & 15)) * lda + k0;
  v8bf a0 = *(const v8bf*)(row + 8 * hf);
  v8bf a1 = *(const v8bf*)(row + 16 + 8 * hf);
  v16bf r;
#pragma unroll
  for (int i = 0; i < 8; ++i) { r[i] = a0[i]; r[i + 8] = a1[i]; }
  return r;
}

// B matrix 32x16 bf16 built from row-major weights Wt[N][K] (B[k][n]=Wt[n][k]):
// lane n = lane%16, K = {k0+8hf..+7, k0+16+8hf..+7} (same order as A).
__device__ __forceinline__ v16bf load_frag_b(const __bf16* Wt, int ldw,
                                             int n0, int k0) {
  const int lane = threadIdx.x & 31;
  const int hf   = lane >> 4;
  const __bf16* p = Wt + (long)(n0 + (lane & 15)) * ldw + k0;
  v8bf b0 = *(const v8bf*)(p + 8 * hf);
  v8bf b1 = *(const v8bf*)(p + 16 + 8 * hf);
  v16bf r;
#pragma unroll
  for (int i = 0; i < 8; ++i) { r[i] = b0[i]; r[i + 8] = b1[i]; }
  return r;
}

// ---- Generic fused GEMM: C = A0*W0^T (+ A1*W1^T) (+ bias) (+ addmat) ------
// Weights row-major [N, K] with ld == K. Block = 128 threads = 4 waves,
// each wave owns a 16(M) x 64(N) strip; block tile = 16 x 256.
__global__ __launch_bounds__(128) void gemm_gates_kernel(
    const __bf16* __restrict__ A0, int lda0, int K0, const __bf16* __restrict__ W0,
    const __bf16* __restrict__ A1, int lda1, int K1, const __bf16* __restrict__ W1w,
    const float* __restrict__ bias, const float* __restrict__ addmat,
    float* __restrict__ C, int ldc) {
  const int wave = threadIdx.x >> 5;
  const int lane = threadIdx.x & 31;
  const int m0 = blockIdx.x * 16;
  const int n0 = blockIdx.y * 256 + wave * 64;

  v8f acc[4];
#pragma unroll
  for (int j = 0; j < 4; ++j)
#pragma unroll
    for (int r = 0; r < 8; ++r) acc[j][r] = 0.f;

  for (int k0 = 0; k0 < K0; k0 += 32) {
    v16bf a = load_frag_a(A0, lda0, m0, k0);
#pragma unroll
    for (int j = 0; j < 4; ++j) {
      v16bf b = load_frag_b(W0, K0, n0 + 16 * j, k0);
      acc[j] = __builtin_amdgcn_wmma_f32_16x16x32_bf16(
          false, a, false, b, (short)0, acc[j], false, false);
    }
  }
  for (int k0 = 0; k0 < K1; k0 += 32) {
    v16bf a = load_frag_a(A1, lda1, m0, k0);
#pragma unroll
    for (int j = 0; j < 4; ++j) {
      v16bf b = load_frag_b(W1w, K1, n0 + 16 * j, k0);
      acc[j] = __builtin_amdgcn_wmma_f32_16x16x32_bf16(
          false, a, false, b, (short)0, acc[j], false, false);
    }
  }

  // C/D layout: lane holds column lane%16, rows 8*(lane/16)+r.
  const int mb = m0 + 8 * (lane >> 4);
#pragma unroll
  for (int j = 0; j < 4; ++j) {
    const int n = n0 + 16 * j + (lane & 15);
    const float bv = bias ? bias[n] : 0.f;
#pragma unroll
    for (int r = 0; r < 8; ++r) {
      float v = acc[j][r] + bv;
      if (addmat) v += addmat[(long)(mb + r) * ldc + n];
      C[(long)(mb + r) * ldc + n] = v;
    }
  }
}

// ---- Pointwise / utility kernels ------------------------------------------

__device__ __forceinline__ float sigmoidf_(float x) {
  return 1.f / (1.f + __expf(-x));
}

__global__ __launch_bounds__(256) void f32_to_bf16_kernel(
    const float* __restrict__ s, __bf16* __restrict__ d, int n) {
  int i = blockIdx.x * 256 + threadIdx.x;
  if (i < n) d[i] = (__bf16)s[i];
}

__global__ __launch_bounds__(256) void add_bias_kernel(
    const float* __restrict__ a, const float* __restrict__ b,
    float* __restrict__ c, int n) {
  int i = blockIdx.x * 256 + threadIdx.x;
  if (i < n) c[i] = a[i] + b[i];
}

__global__ __launch_bounds__(256) void zero_hc_kernel(
    __bf16* __restrict__ Hbf, float* __restrict__ Cst, int n) {
  int i = blockIdx.x * 256 + threadIdx.x;
  if (i < n) { Hbf[i] = (__bf16)0.f; Cst[i] = 0.f; }
}

// Encoder LSTM pointwise: c updated in place; h written as bf16 (recurrence)
// and into enc_outs[t].
__global__ __launch_bounds__(256) void lstm_point_enc_kernel(
    const float* __restrict__ G, float* __restrict__ Cst,
    __bf16* __restrict__ Hbf, __bf16* __restrict__ EncOutT,
    int BH, int Hn, int Gld) {
  int idx = blockIdx.x * 256 + threadIdx.x;
  if (idx >= BH) return;
  int b = idx / Hn, u = idx % Hn;
  const float* g = G + (long)b * Gld;
  float ig = sigmoidf_(g[u]);
  float fg = sigmoidf_(g[u + Hn]);
  float gg = tanhf(g[u + 2 * Hn]);
  float og = sigmoidf_(g[u + 3 * Hn]);
  float c = fg * Cst[idx] + ig * gg;
  float h = og * tanhf(c);
  Cst[idx] = c;
  Hbf[idx] = (__bf16)h;
  EncOutT[idx] = (__bf16)h;
}

// Decoder pointwise: cell is FIXED at encoder final cell (per reference).
__global__ __launch_bounds__(256) void lstm_point_dec_kernel(
    const float* __restrict__ G, const float* __restrict__ Cfix,
    __bf16* __restrict__ Hbf, int BH, int Hn, int Gld) {
  int idx = blockIdx.x * 256 + threadIdx.x;
  if (idx >= BH) return;
  int b = idx / Hn, u = idx % Hn;
  const float* g = G + (long)b * Gld;
  float ig = sigmoidf_(g[u]);
  float fg = sigmoidf_(g[u + Hn]);
  float gg = tanhf(g[u + 2 * Hn]);
  float og = sigmoidf_(g[u + 3 * Hn]);
  float c = fg * Cfix[idx] + ig * gg;
  Hbf[idx] = (__bf16)(og * tanhf(c));
}

// scores[b][s] = sum_w tanh(enc_proj[s][b][w] + d2[b][w]) * vt[w]
// One wave32 per (b,s) pair; enc_proj stays L2-resident (64MB < 192MB L2).
__global__ __launch_bounds__(256) void attn_scores_kernel(
    const float* __restrict__ enc_proj, const float* __restrict__ d2,
    const float* __restrict__ vt, float* __restrict__ scores,
    int B, int S, int Wd) {
  const int wave = threadIdx.x >> 5;
  const int lane = threadIdx.x & 31;
  const int p = blockIdx.x * 8 + wave;  // 8 waves / block
  if (p >= B * S) return;
  const int b = p / S, s = p % S;
  const float* ep = enc_proj + ((long)s * B + b) * Wd;
  const float* dp = d2 + (long)b * Wd;
  float sum = 0.f;
  for (int w = lane; w < Wd; w += 32)
    sum += tanhf(ep[w] + dp[w]) * vt[w];
#pragma unroll
  for (int off = 16; off > 0; off >>= 1)
    sum += __shfl_down(sum, off, 32);
  if (lane == 0) scores[(long)b * S + s] = sum;
}

// log_softmax over S=128 per batch row; one 128-thread block per b.
__global__ __launch_bounds__(128) void logsoftmax_kernel(
    const float* __restrict__ scores, float* __restrict__ out,
    int t, int S, int Sdec) {
  __shared__ float red[128];
  const int b = blockIdx.x, s = threadIdx.x;
  const float v = scores[(long)b * S + s];
  red[s] = v;
  __syncthreads();
  for (int off = 64; off > 0; off >>= 1) {
    if (s < off) red[s] = fmaxf(red[s], red[s + off]);
    __syncthreads();
  }
  const float mx = red[0];
  __syncthreads();
  red[s] = __expf(v - mx);
  __syncthreads();
  for (int off = 64; off > 0; off >>= 1) {
    if (s < off) red[s] += red[s + off];
    __syncthreads();
  }
  const float lse = mx + __logf(red[0]);
  out[((long)b * Sdec + t) * S + s] = v - lse;
}

// ---------------------------------------------------------------------------

extern "C" void kernel_launch(void* const* d_in, const int* in_sizes, int n_in,
                              void* d_out, int out_size, void* d_ws,
                              size_t ws_size, hipStream_t stream) {
  const float* x        = (const float*)d_in[0];
  const float* dec_in   = (const float*)d_in[1];
  const float* W_ih_enc = (const float*)d_in[2];
  const float* W_hh_enc = (const float*)d_in[3];
  const float* b_ih_enc = (const float*)d_in[4];
  const float* b_hh_enc = (const float*)d_in[5];
  const float* W_ih_dec = (const float*)d_in[6];
  const float* W_hh_dec = (const float*)d_in[7];
  const float* b_ih_dec = (const float*)d_in[8];
  const float* b_hh_dec = (const float*)d_in[9];
  const float* W1 = (const float*)d_in[10];
  const float* W2 = (const float*)d_in[11];
  const float* vt = (const float*)d_in[12];
  float* out = (float*)d_out;

  constexpr int B = 256, S = 128, IN = 64, H = 512, Wd = 512;
  constexpr int G = 4 * H;  // 2048
  constexpr int Sdec = S;

  char* ws = (char*)d_ws;
  size_t off = 0;
  auto alloc = [&](size_t bytes) -> char* {
    char* p = ws + off;
    off += (bytes + 255) & ~(size_t)255;
    return p;
  };
  __bf16* Wih_enc_bf = (__bf16*)alloc((size_t)G * IN * 2);
  __bf16* Whh_enc_bf = (__bf16*)alloc((size_t)G * H * 2);
  __bf16* Wih_dec_bf = (__bf16*)alloc((size_t)G * IN * 2);
  __bf16* Whh_dec_bf = (__bf16*)alloc((size_t)G * H * 2);
  __bf16* W1_bf      = (__bf16*)alloc((size_t)Wd * H * 2);
  __bf16* W2_bf      = (__bf16*)alloc((size_t)Wd * H * 2);
  __bf16* x_bf       = (__bf16*)alloc((size_t)B * S * IN * 2);
  __bf16* dec_in_bf  = (__bf16*)alloc((size_t)B * IN * 2);
  float*  b_enc      = (float*)alloc((size_t)G * 4);
  float*  b_dec      = (float*)alloc((size_t)G * 4);
  __bf16* Hbf        = (__bf16*)alloc((size_t)B * H * 2);
  float*  Cst        = (float*)alloc((size_t)B * H * 4);
  __bf16* EncOut     = (__bf16*)alloc((size_t)S * B * H * 2);   // [S,B,H]
  float*  Gates      = (float*)alloc((size_t)B * G * 4);
  float*  enc_proj   = (float*)alloc((size_t)S * B * Wd * 4);   // 64 MB (L2)
  float*  xdec       = (float*)alloc((size_t)B * G * 4);
  float*  d2         = (float*)alloc((size_t)B * Wd * 4);
  float*  scores     = (float*)alloc((size_t)B * S * 4);
  (void)ws_size; (void)in_sizes; (void)n_in; (void)out_size;

  auto cvt = [&](const float* s, __bf16* d, int n) {
    f32_to_bf16_kernel<<<dim3((n + 255) / 256), dim3(256), 0, stream>>>(s, d, n);
  };
  cvt(W_ih_enc, Wih_enc_bf, G * IN);
  cvt(W_hh_enc, Whh_enc_bf, G * H);
  cvt(W_ih_dec, Wih_dec_bf, G * IN);
  cvt(W_hh_dec, Whh_dec_bf, G * H);
  cvt(W1, W1_bf, Wd * H);
  cvt(W2, W2_bf, Wd * H);
  cvt(x, x_bf, B * S * IN);
  cvt(dec_in, dec_in_bf, B * IN);
  add_bias_kernel<<<dim3((G + 255) / 256), dim3(256), 0, stream>>>(
      b_ih_enc, b_hh_enc, b_enc, G);
  add_bias_kernel<<<dim3((G + 255) / 256), dim3(256), 0, stream>>>(
      b_ih_dec, b_hh_dec, b_dec, G);
  zero_hc_kernel<<<dim3((B * H + 255) / 256), dim3(256), 0, stream>>>(
      Hbf, Cst, B * H);

  const dim3 blkG(128);
  const dim3 grdGates(B / 16, G / 256);    // 16 x 8
  const dim3 grdProj(S * B / 16, Wd / 256);// 2048 x 2
  const dim3 grdD2(B / 16, Wd / 256);      // 16 x 2
  const dim3 grdPt((B * H + 255) / 256);
  const dim3 grdAttn(B * S / 8);

  // ---- Encoder: 128 sequential LSTM steps ----
  for (int t = 0; t < S; ++t) {
    gemm_gates_kernel<<<grdGates, blkG, 0, stream>>>(
        x_bf + (size_t)t * IN, S * IN, IN, Wih_enc_bf,
        Hbf, H, H, Whh_enc_bf,
        b_enc, nullptr, Gates, G);
    lstm_point_enc_kernel<<<grdPt, dim3(256), 0, stream>>>(
        Gates, Cst, Hbf, EncOut + (size_t)t * B * H, B * H, H, G);
  }

  // ---- enc_proj = enc_outs @ W1^T : [S*B, 512] x [512, 512] ----
  gemm_gates_kernel<<<grdProj, blkG, 0, stream>>>(
      EncOut, H, H, W1_bf,
      nullptr, 0, 0, nullptr,
      nullptr, nullptr, enc_proj, Wd);

  // ---- Decoder input projection (constant across steps) ----
  gemm_gates_kernel<<<grdGates, blkG, 0, stream>>>(
      dec_in_bf, IN, IN, Wih_dec_bf,
      nullptr, 0, 0, nullptr,
      b_dec, nullptr, xdec, G);

  // ---- Decoder: 128 steps (h continues from encoder h_last in Hbf;
  //      cell fixed at Cst == encoder c_last) ----
  for (int t = 0; t < Sdec; ++t) {
    gemm_gates_kernel<<<grdGates, blkG, 0, stream>>>(
        Hbf, H, H, Whh_dec_bf,
        nullptr, 0, 0, nullptr,
        nullptr, xdec, Gates, G);
    lstm_point_dec_kernel<<<grdPt, dim3(256), 0, stream>>>(
        Gates, Cst, Hbf, B * H, H, G);
    gemm_gates_kernel<<<grdD2, blkG, 0, stream>>>(
        Hbf, H, H, W2_bf,
        nullptr, 0, 0, nullptr,
        nullptr, nullptr, d2, Wd);
    attn_scores_kernel<<<grdAttn, dim3(256), 0, stream>>>(
        enc_proj, d2, vt, scores, B, S, Wd);
    logsoftmax_kernel<<<dim3(B), dim3(128), 0, stream>>>(
        scores, out, t, S, Sdec);
  }
}
